// Informer_55997783605480
// MI455X (gfx1250) — matrix operand order repaired and checked by reading, not compiled, not run on hardware
//
#include <hip/hip_runtime.h>
#include <hip/hip_bf16.h>
#include <math.h>

typedef __attribute__((ext_vector_type(16))) __bf16 v16bf;
typedef __attribute__((ext_vector_type(8)))  float  v8f;

#define B_   8
#define L_   1024
#define C_   32
#define D_   512
#define H_   8
#define DH_  64
#define DFF_ 2048
#define EL_  3
#define NC_  16
#define U_   35

// ---------------------------------------------------------------------------
// CDNA5 async copy helpers (ASYNCcnt-tracked global -> LDS, ISA 15.18.3 / 9.6)
// ---------------------------------------------------------------------------
__device__ __forceinline__ unsigned lds_off_u32(const void* p) {
    // generic LDS address: low 32 bits are the LDS byte offset (ISA 10.2)
    return (unsigned)(unsigned long long)(uintptr_t)p;
}
__device__ __forceinline__ void async_ld_b128(unsigned lds_off, const void* gptr) {
    asm volatile("global_load_async_to_lds_b128 %0, %1, off"
                 :: "v"(lds_off), "v"((unsigned long long)(uintptr_t)gptr)
                 : "memory");
}
__device__ __forceinline__ void wait_async_le4() {
    asm volatile("s_wait_asynccnt 0x4" ::: "memory");
}
__device__ __forceinline__ void wait_async_le0() {
    asm volatile("s_wait_asynccnt 0x0" ::: "memory");
}

// ---------------------------------------------------------------------------
// f32 -> bf16 pack (2048 elems per block)
// ---------------------------------------------------------------------------
__global__ __launch_bounds__(256)
void k_cvt_bf16(const float* __restrict__ src, __bf16* __restrict__ dst)
{
    const size_t i = ((size_t)blockIdx.x * 256 + threadIdx.x) * 8;
    float4 a = *(const float4*)(src + i);
    float4 b = *(const float4*)(src + i + 4);
    union { __bf16 h[8]; uint4 q; } u;
    u.h[0] = (__bf16)a.x; u.h[1] = (__bf16)a.y; u.h[2] = (__bf16)a.z; u.h[3] = (__bf16)a.w;
    u.h[4] = (__bf16)b.x; u.h[5] = (__bf16)b.y; u.h[6] = (__bf16)b.z; u.h[7] = (__bf16)b.w;
    *(uint4*)(dst + i) = u.q;
}

// ---------------------------------------------------------------------------
// bf16 WMMA GEMM: C[M,N] = A[M,K] @ W[N,K]^T + bias, async double-buffered LDS
// fuse: 1 = relu, 2 = +resid.  Cf (f32) and/or Cb (bf16) outputs, nullable.
// block = 256 threads (8 waves), tile 128x128, K-step 32
// ---------------------------------------------------------------------------
union FragBF { uint4 q[2]; v16bf v; };

__global__ __launch_bounds__(256)
void k_gemm_bf16(const __bf16* __restrict__ A, const __bf16* __restrict__ W,
                 const float* __restrict__ bias, const float* __restrict__ resid,
                 float* __restrict__ Cf, __bf16* __restrict__ Cb,
                 int M, int N, int K, int fuse)
{
    __shared__ __align__(16) __bf16 As[2][128][40];   // +8 bf16 row pad
    __shared__ __align__(16) __bf16 Ws[2][128][40];

    const int tid  = threadIdx.x;
    const int lane = tid & 31;
    const int wid  = tid >> 5;
    const int half = lane >> 4;        // ISA 16x16 layout: lanes 0-15 / 16-31
    const int l16  = lane & 15;
    const int wm   = wid & 3;          // wave row group (4 x 32 rows)
    const int wn   = wid >> 2;         // wave col group (2 x 64 cols)
    const int m0   = blockIdx.y * 128;
    const int n0   = blockIdx.x * 128;

    union { v8f v; float f[8]; } acc[2][4];
#pragma unroll
    for (int i = 0; i < 2; i++)
#pragma unroll
        for (int j = 0; j < 4; j++)
#pragma unroll
            for (int t = 0; t < 8; t++) acc[i][j].f[t] = 0.0f;

    const int lr = tid >> 1;           // staging row 0..127
    const int lc = (tid & 1) * 16;     // staging k element offset 0/16

    const unsigned baseA0 = lds_off_u32(&As[0][0][0]) + (unsigned)(lr * 80 + (tid & 1) * 32);
    const unsigned baseA1 = lds_off_u32(&As[1][0][0]) + (unsigned)(lr * 80 + (tid & 1) * 32);
    const unsigned baseW0 = lds_off_u32(&Ws[0][0][0]) + (unsigned)(lr * 80 + (tid & 1) * 32);
    const unsigned baseW1 = lds_off_u32(&Ws[1][0][0]) + (unsigned)(lr * 80 + (tid & 1) * 32);
    const __bf16* pa0 = A + (size_t)(m0 + lr) * K + lc;
    const __bf16* pw0 = W + (size_t)(n0 + lr) * K + lc;

    const int nk = K >> 5;

    // prime tiles 0 and 1 (4 async b128 per wave per tile)
    async_ld_b128(baseA0,      pa0);
    async_ld_b128(baseA0 + 16, pa0 + 8);
    async_ld_b128(baseW0,      pw0);
    async_ld_b128(baseW0 + 16, pw0 + 8);
    async_ld_b128(baseA1,      pa0 + 32);
    async_ld_b128(baseA1 + 16, pa0 + 40);
    async_ld_b128(baseW1,      pw0 + 32);
    async_ld_b128(baseW1 + 16, pw0 + 40);

    for (int it = 0; it < nk; ++it) {
        if (it + 1 < nk) wait_async_le4();   // tile `it` landed; it+1 in flight
        else             wait_async_le0();
        __syncthreads();

        const int pb = it & 1;
        FragBF fa[2], fb[4];
#pragma unroll
        for (int rt = 0; rt < 2; rt++) {     // A 16x32: lanes0-15 K{0..7,16..23}
            const __bf16* p = &As[pb][wm * 32 + rt * 16 + l16][half * 8];
            fa[rt].q[0] = *(const uint4*)(p);
            fa[rt].q[1] = *(const uint4*)(p + 16);
        }
#pragma unroll
        for (int ct = 0; ct < 4; ct++) {     // B 32x16: lanes0-15 K{0..7,16..23}, N=l16
            const __bf16* p = &Ws[pb][wn * 64 + ct * 16 + l16][half * 8];
            fb[ct].q[0] = *(const uint4*)(p);
            fb[ct].q[1] = *(const uint4*)(p + 16);
        }
#pragma unroll
        for (int rt = 0; rt < 2; rt++)
#pragma unroll
            for (int ct = 0; ct < 4; ct++)
                acc[rt][ct].v = __builtin_amdgcn_wmma_f32_16x16x32_bf16(
                    false, fa[rt].v, false, fb[ct].v,
                    (short)0, acc[rt][ct].v, false, false);

        __syncthreads();                     // all waves done with buffer pb
        if (it + 2 < nk) {
            const int k2 = (it + 2) * 32;
            const unsigned da = pb ? baseA1 : baseA0;
            const unsigned dw = pb ? baseW1 : baseW0;
            async_ld_b128(da,      pa0 + k2);
            async_ld_b128(da + 16, pa0 + k2 + 8);
            async_ld_b128(dw,      pw0 + k2);
            async_ld_b128(dw + 16, pw0 + k2 + 8);
        }
    }

    // epilogue: C layout -> VGPR i holds row (half*8 + i), col = l16
#pragma unroll
    for (int rt = 0; rt < 2; rt++) {
#pragma unroll
        for (int ct = 0; ct < 4; ct++) {
            const int col = n0 + wn * 64 + ct * 16 + l16;
            const float bval = bias ? bias[col] : 0.0f;
#pragma unroll
            for (int i = 0; i < 8; i++) {
                const int row = m0 + wm * 32 + rt * 16 + half * 8 + i;
                float v = acc[rt][ct].f[i] + bval;
                if (fuse == 1) v = v > 0.0f ? v : 0.0f;
                if (fuse == 2) v += resid[(size_t)row * N + col];
                if (Cf) Cf[(size_t)row * N + col] = v;
                if (Cb) Cb[(size_t)row * N + col] = (__bf16)v;
            }
        }
    }
}

// ---------------------------------------------------------------------------
// circular conv1d(C->D, k=3) + sinusoidal PE; emits f32 + bf16 mirrors
// ---------------------------------------------------------------------------
__global__ __launch_bounds__(256)
void k_conv_pe(const float* __restrict__ xe, const float* __restrict__ cw,
               float* __restrict__ X, __bf16* __restrict__ Xb)
{
    const int bl = blockIdx.x;               // b*L + l
    const int b = bl >> 10, l = bl & (L_ - 1);
    __shared__ float xs[3][C_];
    const int tid = threadIdx.x;
    if (tid < 96) {
        int j = tid >> 5, c = tid & 31;
        int ll = (l - 1 + j + L_) & (L_ - 1);
        xs[j][c] = xe[((size_t)b * L_ + ll) * C_ + c];
    }
    __syncthreads();
#pragma unroll
    for (int rep = 0; rep < 2; rep++) {
        const int d = tid + rep * 256;
        const float* w = cw + (size_t)d * C_ * 3;
        float s = 0.0f;
#pragma unroll 8
        for (int c = 0; c < C_; c++)
            s += xs[0][c]*w[c*3+0] + xs[1][c]*w[c*3+1] + xs[2][c]*w[c*3+2];
        const int i2 = d >> 1;
        const float div = __expf(-(float)(2 * i2) * (9.210340371976184f / (float)D_));
        const float ang = (float)l * div;
        s += (d & 1) ? __cosf(ang) : __sinf(ang);
        X[(size_t)bl * D_ + d]  = s;
        Xb[(size_t)bl * D_ + d] = (__bf16)s;
    }
}

// ---------------------------------------------------------------------------
// ProbSparse sampled score (deterministic hash RNG)
// ---------------------------------------------------------------------------
__device__ __forceinline__ unsigned hash_u(unsigned x) {
    x ^= x >> 16; x *= 0x7feb352du; x ^= x >> 15; x *= 0x846ca68bu; x ^= x >> 16;
    return x;
}

__global__ __launch_bounds__(256)
void k_mscore(const float* __restrict__ Q, const float* __restrict__ Kt,
              float* __restrict__ MS, int layer)
{
    const int gid = blockIdx.x * 256 + threadIdx.x;   // (b*H+h)*L + l
    const int l  = gid & (L_ - 1);
    const int bh = gid >> 10;
    const int h  = bh & (H_ - 1), b = bh >> 3;
    const float* qr = Q + ((size_t)(b * L_ + l)) * D_ + h * DH_;
    float4 qv[16];
#pragma unroll
    for (int i = 0; i < 16; i++) qv[i] = *(const float4*)(qr + i * 4);
    float mx = -1e30f, sm = 0.0f;
    for (int u = 0; u < U_; u++) {
        const unsigned idx = hash_u((unsigned)(l * 131u + u * 2654435761u
                                               + (unsigned)layer * 0x9E3779B9u)) & (L_ - 1);
        const float* kr = Kt + ((size_t)(b * L_ + (int)idx)) * D_ + h * DH_;
        float dot = 0.0f;
#pragma unroll
        for (int i = 0; i < 16; i++) {
            float4 kv = *(const float4*)(kr + i * 4);
            dot += qv[i].x*kv.x + qv[i].y*kv.y + qv[i].z*kv.z + qv[i].w*kv.w;
        }
        mx = fmaxf(mx, dot); sm += dot;
    }
    MS[gid] = mx - sm * (1.0f / (float)L_);
}

// ---------------------------------------------------------------------------
// per-(b,h) top-U by iterative argmax (tie -> lowest index)
// ---------------------------------------------------------------------------
__global__ __launch_bounds__(256)
void k_topk(const float* __restrict__ MS, int* __restrict__ TOP)
{
    const int bh = blockIdx.x, tid = threadIdx.x;
    __shared__ float sv[L_];
    __shared__ float rv[256];
    __shared__ int   ri[256];
#pragma unroll
    for (int i = 0; i < 4; i++) sv[tid + i*256] = MS[(size_t)bh * L_ + tid + i*256];
    __syncthreads();
    for (int it = 0; it < U_; it++) {
        float best = -1e30f; int bi = L_;
#pragma unroll
        for (int i = 0; i < 4; i++) {
            const int idx = tid + i * 256;
            const float v = sv[idx];
            if (v > best || (v == best && idx < bi)) { best = v; bi = idx; }
        }
        rv[tid] = best; ri[tid] = bi;
        __syncthreads();
        for (int s = 128; s > 0; s >>= 1) {
            if (tid < s) {
                const float v2 = rv[tid+s]; const int i2 = ri[tid+s];
                if (v2 > rv[tid] || (v2 == rv[tid] && i2 < ri[tid])) { rv[tid]=v2; ri[tid]=i2; }
            }
            __syncthreads();
        }
        if (tid == 0) { TOP[bh * U_ + it] = ri[0]; sv[ri[0]] = -1e30f; }
        __syncthreads();
    }
}

// ---------------------------------------------------------------------------
// full softmax attention for the U selected queries -> UPD[b,h,u,64]
// ---------------------------------------------------------------------------
__global__ __launch_bounds__(256)
void k_attn(const float* __restrict__ Q, const float* __restrict__ Kt,
            const float* __restrict__ V, const int* __restrict__ TOP,
            float* __restrict__ UPD)
{
    const int blk = blockIdx.x;           // (b*H+h)*U + u
    const int u = blk % U_, bh = blk / U_;
    const int h = bh & (H_ - 1), b = bh >> 3;
    const int tid = threadIdx.x;
    __shared__ float qs[DH_];
    __shared__ float sc[L_];
    __shared__ float red[256];
    __shared__ float part[4][DH_];
    const int qrow = TOP[bh * U_ + u];
    if (tid < DH_) qs[tid] = Q[((size_t)(b * L_ + qrow)) * D_ + h * DH_ + tid];
    __syncthreads();
#pragma unroll
    for (int i = 0; i < 4; i++) {
        const int kk = tid + i * 256;
        const float* kr = Kt + ((size_t)(b * L_ + kk)) * D_ + h * DH_;
        float dot = 0.0f;
#pragma unroll 16
        for (int e = 0; e < DH_; e += 4) {
            float4 kv = *(const float4*)(kr + e);
            dot += qs[e]*kv.x + qs[e+1]*kv.y + qs[e+2]*kv.z + qs[e+3]*kv.w;
        }
        sc[kk] = dot * 0.125f;            // 1/sqrt(64)
    }
    __syncthreads();
    float mx = -1e30f;
#pragma unroll
    for (int i = 0; i < 4; i++) mx = fmaxf(mx, sc[tid + i*256]);
    red[tid] = mx; __syncthreads();
    for (int s = 128; s > 0; s >>= 1) { if (tid < s) red[tid] = fmaxf(red[tid], red[tid+s]); __syncthreads(); }
    mx = red[0]; __syncthreads();
    float lsum = 0.0f;
#pragma unroll
    for (int i = 0; i < 4; i++) { const int kk = tid + i*256; const float e = __expf(sc[kk] - mx); sc[kk] = e; lsum += e; }
    red[tid] = lsum; __syncthreads();
    for (int s = 128; s > 0; s >>= 1) { if (tid < s) red[tid] += red[tid+s]; __syncthreads(); }
    const float rsum = 1.0f / red[0];
    __syncthreads();
    const int e = tid & (DH_ - 1), p = tid >> 6;
    float acc = 0.0f;
    for (int kk = p * 256; kk < (p + 1) * 256; kk++)
        acc += sc[kk] * V[((size_t)(b * L_ + kk)) * D_ + h * DH_ + e];
    part[p][e] = acc;
    __syncthreads();
    if (tid < DH_)
        UPD[(size_t)blk * DH_ + tid] =
            (part[0][tid] + part[1][tid] + part[2][tid] + part[3][tid]) * rsum;
}

__global__ __launch_bounds__(256)
void k_vmean(const float* __restrict__ V, float* __restrict__ VM)
{
    const int bh = blockIdx.x;
    const int h = bh & (H_ - 1), b = bh >> 3;
    const int tid = threadIdx.x;
    const int e = tid & (DH_ - 1), p = tid >> 6;
    __shared__ float part[4][DH_];
    float acc = 0.0f;
    for (int l = p * 256; l < (p + 1) * 256; l++)
        acc += V[((size_t)(b * L_ + l)) * D_ + h * DH_ + e];
    part[p][e] = acc;
    __syncthreads();
    if (tid < DH_)
        VM[bh * DH_ + tid] = (part[0][tid] + part[1][tid] + part[2][tid] + part[3][tid])
                             * (1.0f / (float)L_);
}

// ctx (bf16) = broadcast v-mean, then scatter selected-query updates
__global__ __launch_bounds__(256)
void k_fill_ctx(const float* __restrict__ VM, __bf16* __restrict__ CTXb)
{
    const size_t i = (size_t)blockIdx.x * 256 + threadIdx.x;  // b*L*D flat
    const int d = (int)(i & (D_ - 1));
    const size_t bl = i >> 9;
    const int b = (int)(bl >> 10);
    CTXb[i] = (__bf16)VM[(b * H_ + (d >> 6)) * DH_ + (d & 63)];
}

__global__ __launch_bounds__(256)
void k_scatter(const float* __restrict__ UPD, const int* __restrict__ TOP,
               __bf16* __restrict__ CTXb)
{
    const int i = blockIdx.x * 256 + threadIdx.x;   // (bh*U+u)*64 + e
    const int e = i & 63, r = i >> 6;
    const int bh = r / U_;
    const int h = bh & 7, b = bh >> 3;
    const int l = TOP[r];
    CTXb[((size_t)(b * L_ + l)) * D_ + h * DH_ + e] = (__bf16)UPD[i];
}

// LayerNorm over D=512; emits f32 + bf16 mirrors
__global__ __launch_bounds__(256)
void k_layernorm(const float* __restrict__ Xin, const float* __restrict__ g,
                 const float* __restrict__ bb, float* __restrict__ Xout,
                 __bf16* __restrict__ Xoutb)
{
    const int row = blockIdx.x, tid = threadIdx.x;
    __shared__ float rs[256], rq[256];
    const float* p = Xin + (size_t)row * D_;
    const float x0 = p[tid], x1 = p[tid + 256];
    rs[tid] = x0 + x1; rq[tid] = x0 * x0 + x1 * x1;
    __syncthreads();
    for (int s = 128; s > 0; s >>= 1) {
        if (tid < s) { rs[tid] += rs[tid+s]; rq[tid] += rq[tid+s]; }
        __syncthreads();
    }
    const float m = rs[0] * (1.0f / (float)D_);
    const float var = rq[0] * (1.0f / (float)D_) - m * m;
    const float rstd = rsqrtf(var + 1e-5f);
    const float o0 = (x0 - m) * rstd * g[tid]       + bb[tid];
    const float o1 = (x1 - m) * rstd * g[tid + 256] + bb[tid + 256];
    Xout[(size_t)row * D_ + tid]        = o0;
    Xout[(size_t)row * D_ + tid + 256]  = o1;
    Xoutb[(size_t)row * D_ + tid]       = (__bf16)o0;
    Xoutb[(size_t)row * D_ + tid + 256] = (__bf16)o1;
}

__global__ __launch_bounds__(256)
void k_final(const float* __restrict__ XF, const float* __restrict__ mark,
             const float* __restrict__ fw, const float* __restrict__ fb,
             float* __restrict__ out)
{
    const int b = blockIdx.x >> 4, nc = blockIdx.x & 15;
    const int tid = threadIdx.x;
    __shared__ float red[256];
    const float* xb = XF + (size_t)b * L_ * D_;
    const float* wb = fw + (size_t)nc * L_ * D_;
    const float* mk = mark + (size_t)b * L_;
    float acc = 0.0f;
    for (int i = tid * 4; i < L_ * D_; i += 256 * 4) {
        float4 xv = *(const float4*)(xb + i);
        float4 wv = *(const float4*)(wb + i);
        const float m = mk[i >> 9];
        acc += m * (xv.x*wv.x + xv.y*wv.y + xv.z*wv.z + xv.w*wv.w);
    }
    red[tid] = acc; __syncthreads();
    for (int s = 128; s > 0; s >>= 1) { if (tid < s) red[tid] += red[tid+s]; __syncthreads(); }
    if (tid == 0) out[b * NC_ + nc] = red[0] + fb[nc];
}

// ---------------------------------------------------------------------------
extern "C" void kernel_launch(void* const* d_in, const int* in_sizes, int n_in,
                              void* d_out, int out_size, void* d_ws, size_t ws_size,
                              hipStream_t stream)
{
    (void)in_sizes; (void)n_in; (void)out_size; (void)ws_size;
    const float* x_enc  = (const float*)d_in[0];
    const float* x_mark = (const float*)d_in[1];
    const float* conv_w = (const float*)d_in[2];
    const float* Wq = (const float*)d_in[3];  const float* bq = (const float*)d_in[4];
    const float* Wk = (const float*)d_in[5];  const float* bk = (const float*)d_in[6];
    const float* Wv = (const float*)d_in[7];  const float* bv = (const float*)d_in[8];
    const float* Wo = (const float*)d_in[9];  const float* bo = (const float*)d_in[10];
    const float* W1 = (const float*)d_in[11]; const float* b1 = (const float*)d_in[12];
    const float* W2 = (const float*)d_in[13]; const float* b2 = (const float*)d_in[14];
    const float* g1 = (const float*)d_in[15]; const float* be1 = (const float*)d_in[16];
    const float* g2 = (const float*)d_in[17]; const float* be2 = (const float*)d_in[18];
    const float* gF = (const float*)d_in[19]; const float* bF = (const float*)d_in[20];
    const float* fc_w = (const float*)d_in[21]; const float* fc_b = (const float*)d_in[22];
    float* out = (float*)d_out;

    char* ws = (char*)d_ws;
    const size_t NXD = (size_t)B_ * L_ * D_;                      // 4M elems
    float*  X    = (float*) ws; ws += NXD * 4;                    // 16 MB
    float*  Qb   = (float*) ws; ws += NXD * 4;
    float*  Kb   = (float*) ws; ws += NXD * 4;
    float*  Vb   = (float*) ws; ws += NXD * 4;
    __bf16* Xb16 = (__bf16*)ws; ws += NXD * 2;                    // 8 MB
    __bf16* CTXb = (__bf16*)ws; ws += NXD * 2;
    __bf16* Hb16 = (__bf16*)ws; ws += (size_t)B_ * L_ * DFF_ * 2; // 32 MB
    __bf16* WqB  = (__bf16*)ws; ws += (size_t)EL_ * D_ * D_ * 2;
    __bf16* WkB  = (__bf16*)ws; ws += (size_t)EL_ * D_ * D_ * 2;
    __bf16* WvB  = (__bf16*)ws; ws += (size_t)EL_ * D_ * D_ * 2;
    __bf16* WoB  = (__bf16*)ws; ws += (size_t)EL_ * D_ * D_ * 2;
    __bf16* W1B  = (__bf16*)ws; ws += (size_t)EL_ * DFF_ * D_ * 2;
    __bf16* W2B  = (__bf16*)ws; ws += (size_t)EL_ * DFF_ * D_ * 2;
    float*  MS   = (float*) ws; ws += (size_t)B_ * H_ * L_ * 4;
    int*    TOP  = (int*)   ws; ws += (size_t)B_ * H_ * U_ * 4;
    float*  UPD  = (float*) ws; ws += (size_t)B_ * H_ * U_ * DH_ * 4;
    float*  VM   = (float*) ws; ws += (size_t)B_ * H_ * DH_ * 4;
    float*  T    = Kb;   // reuse: K dead after attention

    const dim3 blk(256);
    const dim3 g_dd(D_ / 128, (B_ * L_) / 128);     // (4, 64)  N=512 GEMMs
    const dim3 g_df(DFF_ / 128, (B_ * L_) / 128);   // (16, 64) N=2048 GEMM

    // one-time (per launch) weight conversion to bf16
    const size_t NW  = (size_t)EL_ * D_ * D_;       // 786432
    const size_t NW2 = (size_t)EL_ * DFF_ * D_;     // 3145728
    k_cvt_bf16<<<(unsigned)(NW  / 2048), blk, 0, stream>>>(Wq, WqB);
    k_cvt_bf16<<<(unsigned)(NW  / 2048), blk, 0, stream>>>(Wk, WkB);
    k_cvt_bf16<<<(unsigned)(NW  / 2048), blk, 0, stream>>>(Wv, WvB);
    k_cvt_bf16<<<(unsigned)(NW  / 2048), blk, 0, stream>>>(Wo, WoB);
    k_cvt_bf16<<<(unsigned)(NW2 / 2048), blk, 0, stream>>>(W1, W1B);
    k_cvt_bf16<<<(unsigned)(NW2 / 2048), blk, 0, stream>>>(W2, W2B);

    k_conv_pe<<<B_ * L_, blk, 0, stream>>>(x_enc, conv_w, X, Xb16);

    for (int l = 0; l < EL_; l++) {
        const size_t o  = (size_t)l * D_ * D_;
        const size_t o1 = (size_t)l * DFF_ * D_;
        k_gemm_bf16<<<g_dd, blk, 0, stream>>>(Xb16, WqB + o, bq + l * D_, nullptr,
                                              Qb, nullptr, B_*L_, D_, D_, 0);
        k_gemm_bf16<<<g_dd, blk, 0, stream>>>(Xb16, WkB + o, bk + l * D_, nullptr,
                                              Kb, nullptr, B_*L_, D_, D_, 0);
        k_gemm_bf16<<<g_dd, blk, 0, stream>>>(Xb16, WvB + o, bv + l * D_, nullptr,
                                              Vb, nullptr, B_*L_, D_, D_, 0);

        k_mscore<<<(B_ * H_ * L_) / 256, blk, 0, stream>>>(Qb, Kb, MS, l);
        k_topk<<<B_ * H_, blk, 0, stream>>>(MS, TOP);
        k_attn<<<B_ * H_ * U_, blk, 0, stream>>>(Qb, Kb, Vb, TOP, UPD);
        k_vmean<<<B_ * H_, blk, 0, stream>>>(Vb, VM);
        k_fill_ctx<<<(B_ * L_ * D_) / 256, blk, 0, stream>>>(VM, CTXb);
        k_scatter<<<(B_ * H_ * U_ * DH_) / 256, blk, 0, stream>>>(UPD, TOP, CTXb);

        k_gemm_bf16<<<g_dd, blk, 0, stream>>>(CTXb, WoB + o, bo + l * D_, X,
                                              T, nullptr, B_*L_, D_, D_, 2);
        k_layernorm<<<B_ * L_, blk, 0, stream>>>(T, g1 + l * D_, be1 + l * D_, X, Xb16);

        k_gemm_bf16<<<g_df, blk, 0, stream>>>(Xb16, W1B + o1, b1 + l * DFF_, nullptr,
                                              nullptr, Hb16, B_*L_, DFF_, D_, 1);
        k_gemm_bf16<<<g_dd, blk, 0, stream>>>(Hb16, W2B + o1, b2 + l * D_, X,
                                              T, nullptr, B_*L_, D_, DFF_, 2);
        k_layernorm<<<B_ * L_, blk, 0, stream>>>(T, g2 + l * D_, be2 + l * D_, X, Xb16);
    }

    k_layernorm<<<B_ * L_, blk, 0, stream>>>(X, gF, bF, T, CTXb);  // CTXb = dead scratch
    k_final<<<B_ * NC_, blk, 0, stream>>>(T, x_mark, fc_w, fc_b, out);
}
